// attention_block_90744069030545
// MI455X (gfx1250) — matrix operand order repaired and checked by reading, not compiled, not run on hardware
//
#include <hip/hip_runtime.h>

typedef _Float16 half_t;
typedef __attribute__((ext_vector_type(16))) _Float16 v16h;
typedef __attribute__((ext_vector_type(8)))  float    v8f;
typedef int v4i_t __attribute__((vector_size(16)));

#define D_EMBED 128
#define NHEADS  4
#define HDIM    32
#define NTOK    4096
// (1/sqrt(32)) * log2(e): scores come out of WMMA already in log2 domain
#define QK_SCALE (0.17677669529663687f * 1.4426950408889634f)

#if defined(__gfx1250__) && __has_builtin(__builtin_amdgcn_global_load_async_to_lds_b128) && __has_builtin(__builtin_amdgcn_s_wait_asynccnt)
#define USE_ASYNC 1
#else
#define USE_ASYNC 0
#endif

union V16U { uint4 u[2]; v16h h; };

__device__ inline v16h load16(const half_t* p0, const half_t* p1) {
  V16U x;
  x.u[0] = *reinterpret_cast<const uint4*>(p0);
  x.u[1] = *reinterpret_cast<const uint4*>(p1);
  return x.h;
}

// 16-byte global -> LDS copy; async path on gfx1250, reg-staged fallback.
__device__ inline void copy16_g2l(const half_t* g, half_t* l) {
#if USE_ASYNC
  __builtin_amdgcn_global_load_async_to_lds_b128(
      (__attribute__((address_space(1))) v4i_t*)(unsigned long long)g,
      (__attribute__((address_space(3))) v4i_t*)(unsigned int)(unsigned long long)l,
      0, 0);
#else
  *(uint4*)l = *(const uint4*)g;
#endif
}

__device__ inline void wait_async_lds() {
#if USE_ASYNC
  __builtin_amdgcn_s_wait_asynccnt(0);
#endif
}

// ---------------------------------------------------------------------------
// Pass 0: W (128 x 384 f32, row-major) -> Wt (384 x 128 f16, output-major)
// ---------------------------------------------------------------------------
__global__ __launch_bounds__(256) void wt_convert(const float* __restrict__ W,
                                                  half_t* __restrict__ Wt) {
  int i = blockIdx.x * 256 + threadIdx.x;
  if (i < 384 * 128) {
    int m = i >> 7;
    int k = i & 127;
    Wt[i] = (half_t)W[k * 384 + m];
  }
}

// ---------------------------------------------------------------------------
// Pass 1: QKV projection.  One block = 16 tokens; 8 waves cover the 24
// output 16-col tiles.  Q stored [bh][n][32] pre-scaled into the exp2 domain,
// K stored [bh][n][32], V stored transposed [bh][32][n].
// ---------------------------------------------------------------------------
__global__ __launch_bounds__(256) void qkv_wmma(const float* __restrict__ x,
                                                const half_t* __restrict__ Wt,
                                                const float* __restrict__ bias,
                                                half_t* __restrict__ Qh,
                                                half_t* __restrict__ Kh,
                                                half_t* __restrict__ Vt) {
  __shared__ __align__(16) half_t tt[16][136];   // token-major, 16B-aligned rows
  const int blk = blockIdx.x;                    // b*256 + ntile
  const int b   = blk >> 8;
  const int n0  = (blk & 255) << 4;
  const int tid = threadIdx.x;

  // stage x tile with float4 loads: tt[token][channel] = x[b][channel][token]
  for (int e = tid; e < 512; e += 256) {         // 512 float4 = 16 tok x 128 ch
    int k  = e >> 2;                             // channel 0..127
    int nn = (e & 3) << 2;                       // token 0,4,8,12
    float4 xv = *(const float4*)&x[(size_t)b * (D_EMBED * NTOK) +
                                   (size_t)k * NTOK + n0 + nn];
    tt[nn + 0][k] = (half_t)xv.x;
    tt[nn + 1][k] = (half_t)xv.y;
    tt[nn + 2][k] = (half_t)xv.z;
    tt[nn + 3][k] = (half_t)xv.w;
  }
  __syncthreads();

  const int wid  = tid >> 5;
  const int lane = tid & 31;
  const int l16  = lane & 15;
  const int hi   = lane >> 4;
  const int kb8  = hi << 3;
  const int kb16 = hi << 4;

  for (int mt = wid; mt < 24; mt += 8) {
    v8f acc = {};
    const int m = (mt << 4) + l16;               // output column 0..383
    const half_t* wrow = Wt + m * 128;
    for (int kc = 0; kc < 128; kc += 32) {
      v16h aT = load16(&tt[l16][kc + kb8], &tt[l16][kc + 16 + kb8]);
      v16h bW = load16(wrow + kc + kb8, wrow + kc + 16 + kb8);
      acc = __builtin_amdgcn_wmma_f32_16x16x32_f16(false, aT, false, bW,
                                                   (short)0, acc, false, false);
    }
    float bv   = bias[m];
    int   qkv  = m >> 7;
    int   cc   = m & 127;
    int   head = cc >> 5;
    int   d    = cc & 31;
    size_t bh  = (size_t)(b * NHEADS + head);
#pragma unroll
    for (int v = 0; v < 8; ++v) {
      int   n   = n0 + (hi << 3) + v;
      float val = acc[v] + bv;
      if (qkv == 0)      Qh[(bh * NTOK + n) * HDIM + d] = (half_t)(val * QK_SCALE);
      else if (qkv == 1) Kh[(bh * NTOK + n) * HDIM + d] = (half_t)val;
      else               Vt[(bh * HDIM + d) * NTOK + n] = (half_t)val;
    }
  }
}

// ---------------------------------------------------------------------------
// Pass 2: block-cooperative flash attention.  One block = 128 queries of one
// (b,head); 8 waves each own a 16-query tile.  K/V tiles (32 keys) staged to
// LDS once per block via async-to-LDS, double-buffered.
// ---------------------------------------------------------------------------
__device__ inline void stage_kv(const half_t* Kbase, const half_t* Vbase, int kv,
                                half_t (*Kt)[32], half_t (*Vt)[32], int tid) {
  const int t   = tid & 127;
  const int row = t >> 2;          // 0..31 (key index for K, d index for V)
  const int seg = (t & 3) << 3;    // 0,8,16,24 halves (16B segments)
  if (tid < 128)
    copy16_g2l(Kbase + (size_t)(kv + row) * HDIM + seg, &Kt[row][seg]);
  else
    copy16_g2l(Vbase + (size_t)row * NTOK + kv + seg, &Vt[row][seg]);
}

__global__ __launch_bounds__(256) void attn_wmma(const float* __restrict__ x,
                                                 const half_t* __restrict__ Qh,
                                                 const half_t* __restrict__ Kh,
                                                 const half_t* __restrict__ Vt,
                                                 float* __restrict__ out) {
  __shared__ __align__(16) half_t Ktile[2][32][32];   // [buf][key][d]
  __shared__ __align__(16) half_t Vtile[2][32][32];   // [buf][d][key]
  __shared__ __align__(16) half_t Pbuf[8][16 * 32];   // per-wave private

  const int tid  = threadIdx.x;
  const int wid  = tid >> 5;
  const int lane = tid & 31;
  const int l16  = lane & 15;
  const int hi   = lane >> 4;
  const int kb8  = hi << 3;
  const int kb16 = hi << 4;

  const int bid  = blockIdx.x;          // b*128 + head*32 + qb   (512 blocks)
  const int qb   = bid & 31;
  const int head = (bid >> 5) & 3;
  const int b    = bid >> 7;
  const size_t bh = (size_t)(b * NHEADS + head);
  const int q0   = qb * 128 + wid * 16;

  const half_t* Qbase = Qh + (bh * NTOK + q0) * HDIM;
  const half_t* Kbase = Kh + bh * NTOK * HDIM;
  const half_t* Vbase = Vt + bh * HDIM * NTOK;
  half_t* Pw = &Pbuf[wid][0];

  v16h aQ = load16(Qbase + l16 * HDIM + kb8,
                   Qbase + l16 * HDIM + 16 + kb8);

  float mrow[8], lrow[8];
  v8f O0 = {}, O1 = {};
#pragma unroll
  for (int v = 0; v < 8; ++v) { mrow[v] = -1e30f; lrow[v] = 0.f; }

  stage_kv(Kbase, Vbase, 0, Ktile[0], Vtile[0], tid);

  for (int kv = 0; kv < NTOK; kv += 32) {
    const int cur = (kv >> 5) & 1;
    wait_async_lds();
    __syncthreads();
    if (kv + 32 < NTOK)
      stage_kv(Kbase, Vbase, kv + 32, Ktile[cur ^ 1], Vtile[cur ^ 1], tid);

    v16h bK0 = load16(&Ktile[cur][l16][kb8],      &Ktile[cur][l16][16 + kb8]);
    v16h bK1 = load16(&Ktile[cur][16 + l16][kb8], &Ktile[cur][16 + l16][16 + kb8]);
    v8f z = {};
    v8f s0 = __builtin_amdgcn_wmma_f32_16x16x32_f16(false, aQ, false, bK0,
                                                    (short)0, z, false, false);
    v8f s1 = __builtin_amdgcn_wmma_f32_16x16x32_f16(false, aQ, false, bK1,
                                                    (short)0, z, false, false);

#pragma unroll
    for (int v = 0; v < 8; ++v) {
      // scores are already in log2 units (scale folded into Q)
      float t = fmaxf(s0[v], s1[v]);
      t = fmaxf(t, __shfl_xor(t, 1, 32));
      t = fmaxf(t, __shfl_xor(t, 2, 32));
      t = fmaxf(t, __shfl_xor(t, 4, 32));
      t = fmaxf(t, __shfl_xor(t, 8, 32));
      float newm = fmaxf(mrow[v], t);
      float corr = exp2f(mrow[v] - newm);
      mrow[v] = newm;
      float p0 = exp2f(s0[v] - newm);
      float p1 = exp2f(s1[v] - newm);
      float rs = p0 + p1;
      rs += __shfl_xor(rs, 1, 32);
      rs += __shfl_xor(rs, 2, 32);
      rs += __shfl_xor(rs, 4, 32);
      rs += __shfl_xor(rs, 8, 32);
      lrow[v] = lrow[v] * corr + rs;
      O0[v] *= corr;
      O1[v] *= corr;
      int row = (hi << 3) + v;
      Pw[row * 32 + l16]      = (half_t)p0;
      Pw[row * 32 + 16 + l16] = (half_t)p1;
    }

    v16h aP  = load16(Pw + l16 * 32 + kb8, Pw + l16 * 32 + 16 + kb8);
    v16h bV0 = load16(&Vtile[cur][l16][kb8],      &Vtile[cur][l16][16 + kb8]);
    v16h bV1 = load16(&Vtile[cur][16 + l16][kb8], &Vtile[cur][16 + l16][16 + kb8]);
    O0 = __builtin_amdgcn_wmma_f32_16x16x32_f16(false, aP, false, bV0,
                                                (short)0, O0, false, false);
    O1 = __builtin_amdgcn_wmma_f32_16x16x32_f16(false, aP, false, bV1,
                                                (short)0, O1, false, false);
  }

  // normalize, fuse the reference's raw reshape (token-major flat) + residual
#pragma unroll
  for (int v = 0; v < 8; ++v) {
    float inv = 1.0f / lrow[v];
    int   n   = q0 + (hi << 3) + v;
    size_t basei = (size_t)b * (D_EMBED * NTOK) + (size_t)n * D_EMBED + head * HDIM;
    size_t i0 = basei + l16;
    size_t i1 = basei + 16 + l16;
    out[i0] = O0[v] * inv + x[i0];
    out[i1] = O1[v] * inv + x[i1];
  }
}

// ---------------------------------------------------------------------------
extern "C" void kernel_launch(void* const* d_in, const int* in_sizes, int n_in,
                              void* d_out, int out_size, void* d_ws, size_t ws_size,
                              hipStream_t stream) {
  const float* x    = (const float*)d_in[0];
  const float* W    = (const float*)d_in[1];
  const float* bias = (const float*)d_in[2];
  float* out        = (float*)d_out;

  char* ws = (char*)d_ws;
  const size_t WT_BYTES  = (size_t)384 * 128 * sizeof(half_t);                 // 96 KB
  const size_t QKV_BYTES = (size_t)4 * NHEADS * NTOK * HDIM * sizeof(half_t);  // 4 MiB each
  half_t* Wt = (half_t*)ws;
  half_t* Qh = (half_t*)(ws + WT_BYTES);
  half_t* Kh = (half_t*)(ws + WT_BYTES + QKV_BYTES);
  half_t* Vt = (half_t*)(ws + WT_BYTES + 2 * QKV_BYTES);

  wt_convert<<<192, 256, 0, stream>>>(W, Wt);
  qkv_wmma<<<4 * (NTOK / 16), 256, 0, stream>>>(x, Wt, bias, Qh, Kh, Vt);
  attn_wmma<<<4 * NHEADS * (NTOK / 128), 256, 0, stream>>>(x, Qh, Kh, Vt, out);
}
